// CausalSelfAttention_46119358825187
// MI455X (gfx1250) — matrix operand order repaired and checked by reading, not compiled, not run on hardware
//
#include <hip/hip_runtime.h>

typedef __attribute__((ext_vector_type(16))) __bf16 v16bf;
typedef __attribute__((ext_vector_type(8)))  __bf16 v8bf;
typedef __attribute__((ext_vector_type(4)))  __bf16 v4bf;
typedef __attribute__((ext_vector_type(8)))  float  v8f;

#define B_ 2
#define T_ 2048
#define D_ 512
#define H_ 8

#define WMMA_BF16(A, Bf, C) \
    __builtin_amdgcn_wmma_f32_16x16x32_bf16(false, (A), false, (Bf), (short)0, (C), false, false)

// ---------------------------------------------------------------- kernel 1
// h (fp32) -> hb (bf16), 4 elems/thread
__global__ void k_cvt_h(const float* __restrict__ h, __bf16* __restrict__ hb) {
    int i = (blockIdx.x * blockDim.x + threadIdx.x) * 4;
    float4 f = *(const float4*)(h + i);
    v4bf o;
    o[0] = (__bf16)f.x; o[1] = (__bf16)f.y; o[2] = (__bf16)f.z; o[3] = (__bf16)f.w;
    *(v4bf*)(hb + i) = o;
}

// ---------------------------------------------------------------- kernel 2
// A[h][e][n] (fp32) -> Atb[h][n][e] (bf16), LDS-tiled 64x64 transpose.
__global__ void k_cvtT_A(const float* __restrict__ A, __bf16* __restrict__ Atb) {
    __shared__ __bf16 tile[64][72];
    int blk = blockIdx.x;                     // H * 8 * 8 blocks
    int hh  = blk >> 6;
    int eT  = ((blk >> 3) & 7) << 6;
    int nT  = (blk & 7) << 6;
    int tx  = threadIdx.x & 63;
    int ty  = threadIdx.x >> 6;               // 0..3
    const float* src = A + (size_t)hh * D_ * D_;
#pragma unroll
    for (int it = 0; it < 16; ++it) {
        int e = ty + 4 * it;
        tile[e][tx] = (__bf16)src[(size_t)(eT + e) * D_ + nT + tx];
    }
    __syncthreads();
    __bf16* dst = Atb + (size_t)hh * D_ * D_;
#pragma unroll
    for (int it = 0; it < 16; ++it) {
        int n = ty + 4 * it;
        dst[(size_t)(nT + n) * D_ + eT + tx] = tile[tx][n];
    }
}

// Assemble 16-element bf16 A-fragment (ISA 16-bit A 16x32 layout) from two
// contiguous 8-element runs: K = k0+8*half .. +7 and K = k0+16+8*half .. +7.
__device__ __forceinline__ v16bf frag_from_runs(const __bf16* p0, const __bf16* p1) {
    v8bf lo = *(const v8bf*)p0;
    v8bf hi = *(const v8bf*)p1;
    return __builtin_shufflevector(lo, hi, 0,1,2,3,4,5,6,7,8,9,10,11,12,13,14,15);
}

// ---------------------------------------------------------------- kernel 3
// G[bh] = hb[b] @ A[hh].  Block = 8 waves, all sharing the same 16 h-rows:
// the 16x512 bf16 A-tile is staged in LDS once per block (8x A-traffic cut);
// each wave computes a 16x64 N-slice with 4 accumulators, K-loop
// double-buffered (B-frags from global, A-frags from LDS).
__global__ void k_gemm_hA(const __bf16* __restrict__ hb,
                          const __bf16* __restrict__ Atb,
                          __bf16* __restrict__ Gb) {
    __shared__ __bf16 sA[16 * D_];            // 16 KB

    int lane = threadIdx.x & 31;
    int w    = threadIdx.x >> 5;              // wave 0..7 -> nCol group
    int half = lane >> 4, l15 = lane & 15;

    int bh   = blockIdx.x >> 7;               // 128 blocks per (b,h)
    int tRow = (blockIdx.x & 127) << 4;
    int nCol = w << 6;
    int b    = bh >> 3, hh = bh & 7;

    // cooperative load: 16 rows x 512 bf16; wave w loads rows 2w, 2w+1
    const __bf16* hrows = hb + ((size_t)(b * T_) + tRow) * D_;
#pragma unroll
    for (int i = 0; i < 2; ++i) {
        int rl = 2 * w + i;
        *(v16bf*)&sA[rl * D_ + lane * 16] =
            *(const v16bf*)(hrows + (size_t)rl * D_ + lane * 16);
    }
    __syncthreads();

    const __bf16* arow = &sA[l15 * D_];                               // A row l15
    const __bf16* brow = Atb + ((size_t)hh * D_ + nCol + l15) * D_;   // B col l15

    v8f c0 = {}, c1 = {}, c2 = {}, c3 = {};

    int koh = 16 * half;
    v16bf af = *(const v16bf*)(arow + koh);
    v16bf b0 = *(const v16bf*)(brow + koh);
    v16bf b1 = *(const v16bf*)(brow + 16 * D_ + koh);
    v16bf b2 = *(const v16bf*)(brow + 32 * D_ + koh);
    v16bf b3 = *(const v16bf*)(brow + 48 * D_ + koh);

#pragma unroll 3
    for (int k0 = 0; k0 < D_ - 32; k0 += 32) {
        int kn = k0 + 32, ko = kn + koh;
        v16bf afn = *(const v16bf*)(arow + ko);
        v16bf b0n = *(const v16bf*)(brow + ko);
        v16bf b1n = *(const v16bf*)(brow + 16 * D_ + ko);
        v16bf b2n = *(const v16bf*)(brow + 32 * D_ + ko);
        v16bf b3n = *(const v16bf*)(brow + 48 * D_ + ko);
        c0 = WMMA_BF16(af, b0, c0);
        c1 = WMMA_BF16(af, b1, c1);
        c2 = WMMA_BF16(af, b2, c2);
        c3 = WMMA_BF16(af, b3, c3);
        af = afn; b0 = b0n; b1 = b1n; b2 = b2n; b3 = b3n;
    }
    c0 = WMMA_BF16(af, b0, c0);
    c1 = WMMA_BF16(af, b1, c1);
    c2 = WMMA_BF16(af, b2, c2);
    c3 = WMMA_BF16(af, b3, c3);

    // C/D layout: lane%16 = N, VGPR j = row j + 8*(lane/16)
    __bf16* obase = Gb + ((size_t)bh * T_ + tRow) * D_ + nCol + l15;
#pragma unroll
    for (int j = 0; j < 8; ++j) {
        int M = j + half * 8;
        __bf16* p = obase + (size_t)M * D_;
        p[0]  = (__bf16)c0[j];
        p[16] = (__bf16)c1[j];
        p[32] = (__bf16)c2[j];
        p[48] = (__bf16)c3[j];
    }
}

__device__ __forceinline__ float maskv(float v, int si, int tBlk, int l15, int M) {
    return (si > tBlk || (si == tBlk && l15 > M)) ? -__builtin_inff() : v;
}

// ---------------------------------------------------------------- kernel 4
// Causal scores S = G_rows @ hb^T with online softmax -> diagonal weight.
// Block = 8 waves owning 128 query rows of one (b,h); waves walk the 64-wide
// key groups in lockstep so each 64x512 bf16 key tile is staged in LDS ONCE
// per block and reused by all 8 waves (8x key-traffic cut).
__global__ void k_scores_diag(const __bf16* __restrict__ hb,
                              const __bf16* __restrict__ Gb,
                              float* __restrict__ diag) {
    __shared__ __bf16 sK[64 * D_];            // 64 KB key tile

    int lane = threadIdx.x & 31;
    int w    = threadIdx.x >> 5;              // wave 0..7
    int half = lane >> 4, l15 = lane & 15;

    int bh   = blockIdx.x >> 4;               // T/128 = 16 blocks per (b,h)
    int tb   = (blockIdx.x & 15) << 7;        // block's first query row
    int tBlk = tb + (w << 4);                 // this wave's 16-row strip
    int b    = bh >> 3;

    const __bf16* grow  = Gb + ((size_t)bh * T_ + tBlk + l15) * D_;   // A row l15
    const __bf16* hbase = hb + (size_t)b * T_ * D_;

    float m[8], l[8], dsc[8];
#pragma unroll
    for (int j = 0; j < 8; ++j) { m[j] = -__builtin_inff(); l[j] = 0.f; dsc[j] = 0.f; }

    int sgEnd = tb + 64;                      // last group start (covers tb+112)
    for (int sg = 0; sg <= sgEnd; sg += 64) {
        // ---- cooperative stage: rows sg..sg+63 -> LDS (1 KB per wave-row)
        __syncthreads();                      // protect previous group's readers
#pragma unroll
        for (int i = 0; i < 8; ++i) {
            int rl = w + 8 * i;               // 0..63
            *(v16bf*)&sK[rl * D_ + lane * 16] =
                *(const v16bf*)(hbase + (size_t)(sg + rl) * D_ + lane * 16);
        }
        __syncthreads();
        if (sg + 64 <= sgEnd)                 // prefetch next group's rows
            __builtin_prefetch(hbase + (size_t)(sg + 64 + w * 8) * D_, 0, 0);

        if (sg > tBlk) continue;              // beyond this wave's causal limit

        v8f a0 = {}, a1 = {}, a2 = {}, a3 = {};

        int koh = 16 * half;
        v16bf af = *(const v16bf*)(grow + koh);
        v16bf f0 = *(const v16bf*)&sK[l15 * D_ + koh];
        v16bf f1 = *(const v16bf*)&sK[(16 + l15) * D_ + koh];
        v16bf f2 = *(const v16bf*)&sK[(32 + l15) * D_ + koh];
        v16bf f3 = *(const v16bf*)&sK[(48 + l15) * D_ + koh];

#pragma unroll 3
        for (int k0 = 0; k0 < D_ - 32; k0 += 32) {
            int kn = k0 + 32, ko = kn + koh;
            v16bf afn = *(const v16bf*)(grow + ko);
            v16bf f0n = *(const v16bf*)&sK[l15 * D_ + ko];
            v16bf f1n = *(const v16bf*)&sK[(16 + l15) * D_ + ko];
            v16bf f2n = *(const v16bf*)&sK[(32 + l15) * D_ + ko];
            v16bf f3n = *(const v16bf*)&sK[(48 + l15) * D_ + ko];
            a0 = WMMA_BF16(af, f0, a0);
            a1 = WMMA_BF16(af, f1, a1);
            a2 = WMMA_BF16(af, f2, a2);
            a3 = WMMA_BF16(af, f3, a3);
            af = afn; f0 = f0n; f1 = f1n; f2 = f2n; f3 = f3n;
        }
        a0 = WMMA_BF16(af, f0, a0);
        a1 = WMMA_BF16(af, f1, a1);
        a2 = WMMA_BF16(af, f2, a2);
        a3 = WMMA_BF16(af, f3, a3);

        int di = (tBlk - sg) >> 4;            // diag sub-tile index if < 4
#pragma unroll
        for (int j = 0; j < 8; ++j) {
            int M = j + half * 8;             // row index within the 16-row strip
            float v0 = maskv(a0[j], sg,      tBlk, l15, M);
            float v1 = maskv(a1[j], sg + 16, tBlk, l15, M);
            float v2 = maskv(a2[j], sg + 32, tBlk, l15, M);
            float v3 = maskv(a3[j], sg + 48, tBlk, l15, M);

            float tm = fmaxf(fmaxf(v0, v1), fmaxf(v2, v3));
            for (int msk = 1; msk < 16; msk <<= 1) tm = fmaxf(tm, __shfl_xor(tm, msk, 32));
            float nm = fmaxf(m[j], tm);
            float s = __expf(v0 - nm) + __expf(v1 - nm) + __expf(v2 - nm) + __expf(v3 - nm);
            for (int msk = 1; msk < 16; msk <<= 1) s += __shfl_xor(s, msk, 32);
            l[j] = l[j] * __expf(m[j] - nm) + s;
            m[j] = nm;

            if (di < 4) {                     // diagonal lives in this 64-group
                float dv = (di == 0) ? v0 : (di == 1) ? v1 : (di == 2) ? v2 : v3;
                dv = (l15 == M) ? dv : -__builtin_inff();
                for (int msk = 1; msk < 16; msk <<= 1) dv = fmaxf(dv, __shfl_xor(dv, msk, 32));
                dsc[j] = dv;
            }
        }
    }

    if (l15 == 0) {
#pragma unroll
        for (int j = 0; j < 8; ++j) {
            int M = j + half * 8;
            diag[(size_t)bh * T_ + tBlk + M] = __expf(dsc[j] - m[j]) / l[j];
        }
    }
}

// ---------------------------------------------------------------- kernel 5
// out flat (B,H,T,d) row-major == reference reshape(B,T,H*d).
__global__ void k_scale_out(const float* __restrict__ h,
                            const float* __restrict__ diag,
                            float* __restrict__ out) {
    size_t o  = ((size_t)blockIdx.x * blockDim.x + threadIdx.x) * 4;
    int c  = (int)(o % D_);
    int t  = (int)((o / D_) % T_);
    int hh = (int)((o / ((size_t)D_ * T_)) % H_);
    int b  = (int)( o / ((size_t)D_ * T_ * H_));
    float w = diag[(size_t)(b * H_ + hh) * T_ + t];
    float4 hv = *(const float4*)(h + (size_t)(b * T_ + t) * D_ + c);
    float4 ov = make_float4(w * hv.x, w * hv.y, w * hv.z, w * hv.w);
    *(float4*)(out + o) = ov;
}

// ----------------------------------------------------------------
extern "C" void kernel_launch(void* const* d_in, const int* in_sizes, int n_in,
                              void* d_out, int out_size, void* d_ws, size_t ws_size,
                              hipStream_t stream) {
    const float* h = (const float*)d_in[0];   // (B,T,D)
    const float* A = (const float*)d_in[1];   // (H,D,D)
    float* out = (float*)d_out;               // (B,T,H*D) == flat (B,H,T,D)

    char* ws = (char*)d_ws;
    const size_t hb_bytes = (size_t)B_ * T_ * D_ * 2;        // 4 MB
    const size_t at_bytes = (size_t)H_ * D_ * D_ * 2;        // 4 MB
    const size_t gb_bytes = (size_t)B_ * H_ * T_ * D_ * 2;   // 33.5 MB
    __bf16* hb   = (__bf16*)(ws);
    __bf16* Atb  = (__bf16*)(ws + hb_bytes);
    __bf16* Gb   = (__bf16*)(ws + hb_bytes + at_bytes);
    float*  diag = (float*)(ws + hb_bytes + at_bytes + gb_bytes);

    // 1) h -> bf16
    k_cvt_h<<<(B_ * T_ * D_ / 4) / 256, 256, 0, stream>>>(h, hb);
    // 2) A -> bf16 transposed [h][n][e]  (H * 8 * 8 = 512 tiles)
    k_cvtT_A<<<H_ * 8 * 8, 256, 0, stream>>>(A, Atb);
    // 3) G = h @ A : B*H*(T/16) blocks of 8 waves (full N per block)
    k_gemm_hA<<<B_ * H_ * (T_ / 16), 256, 0, stream>>>(hb, Atb, Gb);
    // 4) causal scores + online softmax -> diag : B*H*(T/128) blocks
    k_scores_diag<<<B_ * H_ * (T_ / 128), 256, 0, stream>>>(hb, Gb, diag);
    // 5) out = diag * h
    k_scale_out<<<(B_ * H_ * T_ * D_ / 4) / 256, 256, 0, stream>>>(h, diag, out);
}
